// GAT_15625091022897
// MI455X (gfx1250) — matrix operand order repaired and checked by reading, not compiled, not run on hardware
//
#include <hip/hip_runtime.h>
#include <math.h>

typedef __attribute__((ext_vector_type(16))) _Float16 v16h;
typedef __attribute__((ext_vector_type(8)))  float    v8f;

#define F_IN   256
#define HID    16
#define HEADS  8
#define H1DIM  128          // HEADS*HID
#define CLS    40
#define NEG_SLOPE 0.2f
#define GAT_EPS 1e-16f

// ---------------- helpers ----------------
__device__ __forceinline__ unsigned fenc(float f) {
    unsigned u = __float_as_uint(f);
    return (u & 0x80000000u) ? ~u : (u | 0x80000000u);
}
__device__ __forceinline__ float fdec(unsigned u) {
    return __uint_as_float((u & 0x80000000u) ? (u ^ 0x80000000u) : ~u);
}
__device__ __forceinline__ void edge_nodes(int e, const int* __restrict__ ei, int E,
                                           int& s, int& d) {
    if (e < E) { s = ei[e]; d = ei[E + e]; }
    else       { s = e - E; d = e - E; }      // self loops appended
}
__device__ __forceinline__ float leaky(float v) {
    return v > 0.f ? v : NEG_SLOPE * v;
}

// Async global -> LDS copy of 16 bytes (CDNA5 GLOBAL_LOAD_ASYNC_TO_LDS_B128).
// lds_off: wave-relative LDS byte address (low 32 bits of shared flat address).
__device__ __forceinline__ void async_copy_b128(unsigned lds_off, const float* gaddr) {
    asm volatile("global_load_async_to_lds_b128 %0, %1, off"
                 :: "v"(lds_off), "v"(gaddr)
                 : "memory");
}
__device__ __forceinline__ void async_wait0() {
    asm volatile("s_wait_asynccnt 0x0" ::: "memory");
}

// ---------------- GEMM1: h1[N,128] = X[N,256] @ W1[256,128] (WMMA f16->f32) ----
// A tile staged f32 in LDS via async-to-LDS DMA; converted to f16 in fragments.
#define A1_STRIDE (F_IN + 4)   // 260 floats: keeps 16B alignment, spreads banks
__global__ __launch_bounds__(256) void gat_gemm1_wmma(
        const float* __restrict__ X, const float* __restrict__ W,
        float* __restrict__ Hout, int N) {
    __shared__ float Asf[16][A1_STRIDE];
    const int rowBase = blockIdx.x * 16;
    const unsigned ldsBase = (unsigned)(uintptr_t)&Asf[0][0];
    // 16 rows x 256 cols = 1024 float4 groups, 256 threads -> 4 async copies each
    for (int i = threadIdx.x; i < 16 * (F_IN / 4); i += 256) {
        int r = i >> 6, g = i & 63;
        int rr = rowBase + r; if (rr >= N) rr = N - 1;
        async_copy_b128(ldsBase + (unsigned)(r * A1_STRIDE + g * 4) * 4u,
                        X + (size_t)rr * F_IN + g * 4);
    }
    async_wait0();
    __syncthreads();

    const int lane = threadIdx.x & 31;
    const int wave = threadIdx.x >> 5;        // 8 waves -> 8 heads (16 cols each)
    const int colBase = wave * 16;
    const int m    = lane & 15;
    const int ksel = lane >> 4;
    v8f acc = {};
    for (int k0 = 0; k0 < F_IN; k0 += 32) {
        v16h a, b;
        const int ka = k0 + ksel * 8;         // A: lanes0-15 K=[k0..k0+7],[k0+16..23]
        #pragma unroll
        for (int i = 0; i < 8; ++i) {
            a[i]     = (_Float16)Asf[m][ka + i];
            a[8 + i] = (_Float16)Asf[m][ka + 16 + i];
        }
        const int kb = k0 + ksel * 8;         // B: same K order as A
        const float* Wc = W + (colBase + m);
        #pragma unroll
        for (int i = 0; i < 16; ++i)
            b[i] = (_Float16)Wc[(size_t)(kb + (i & 7) + ((i >> 3) << 4)) * H1DIM];
        acc = __builtin_amdgcn_wmma_f32_16x16x32_f16(false, a, false, b,
                                                     (short)0, acc, false, false);
    }
    const int mb = ksel * 8;                  // C: vgpr j -> row mb+j, col = lane&15
    if (rowBase + 16 <= N) {                  // full tile: unguarded coalesced stores
        float* outp = Hout + (size_t)(rowBase + mb) * H1DIM + colBase + m;
        #pragma unroll
        for (int j = 0; j < 8; ++j) outp[(size_t)j * H1DIM] = acc[j];
    } else {
        #pragma unroll
        for (int j = 0; j < 8; ++j) {
            int r = rowBase + mb + j;
            if (r < N) Hout[(size_t)r * H1DIM + colBase + m] = acc[j];
        }
    }
}

// ---------------- GEMM2: x2[N,40] = Helu[N,128] @ W2[128,40] ----------------
__global__ __launch_bounds__(96) void gat_gemm2_wmma(
        const float* __restrict__ Hin, const float* __restrict__ W,
        float* __restrict__ Out, int N) {
    __shared__ _Float16 As[16][H1DIM + 8];
    const int rowBase = blockIdx.x * 16;
    for (int i = threadIdx.x; i < 16 * H1DIM; i += 96) {
        int r = i >> 7, c = i & (H1DIM - 1);
        int rr = rowBase + r; if (rr >= N) rr = N - 1;
        As[r][c] = (_Float16)Hin[(size_t)rr * H1DIM + c];
    }
    __syncthreads();
    const int lane = threadIdx.x & 31;
    const int wave = threadIdx.x / 32;        // 3 waves cover 48 cols (40 used)
    const int colBase = wave * 16;
    const int m    = lane & 15;
    const int ksel = lane >> 4;
    const int col  = colBase + m;
    const bool colOK = (col < CLS);
    v8f acc = {};
    for (int k0 = 0; k0 < H1DIM; k0 += 32) {
        v16h a, b;
        const int ka = k0 + ksel * 8;
        #pragma unroll
        for (int i = 0; i < 8; ++i) {
            a[i]     = As[m][ka + i];
            a[8 + i] = As[m][ka + 16 + i];
        }
        const int kb = k0 + ksel * 8;
        #pragma unroll
        for (int i = 0; i < 16; ++i)
            b[i] = colOK ? (_Float16)W[(size_t)(kb + (i & 7) + ((i >> 3) << 4)) * CLS + col] : (_Float16)0.f;
        acc = __builtin_amdgcn_wmma_f32_16x16x32_f16(false, a, false, b,
                                                     (short)0, acc, false, false);
    }
    if (colOK) {
        const int mb = ksel * 8;
        if (rowBase + 16 <= N) {
            float* outp = Out + (size_t)(rowBase + mb) * CLS + col;
            #pragma unroll
            for (int j = 0; j < 8; ++j) outp[(size_t)j * CLS] = acc[j];
        } else {
            #pragma unroll
            for (int j = 0; j < 8; ++j) {
                int r = rowBase + mb + j;
                if (r < N) Out[(size_t)r * CLS + col] = acc[j];
            }
        }
    }
}

// ---------------- attention coefficients: a_s/a_d = <h[n,h,:], att> ----------
__global__ void gat_attn_coeff(const float* __restrict__ Hmat,
                               const float* __restrict__ att_src,
                               const float* __restrict__ att_dst,
                               float* __restrict__ as_, float* __restrict__ ad_,
                               int N, int heads, int outch) {
    int idx = blockIdx.x * blockDim.x + threadIdx.x;
    if (idx >= N * heads) return;
    int n = idx / heads, h = idx - n * heads;
    const float* hp = Hmat + ((size_t)n * heads + h) * outch;
    const float* s  = att_src + h * outch;
    const float* d  = att_dst + h * outch;
    float ss = 0.f, dd = 0.f;
    for (int c = 0; c < outch; ++c) { float v = hp[c]; ss += v * s[c]; dd += v * d[c]; }
    as_[idx] = ss;
    ad_[idx] = dd;
}

// ---------------- edge pass A: segment max (monotonic u32 encoding) ----------
template <int H>
__global__ void gat_edge_max(const int* __restrict__ ei, int E, int Etot,
                             const float* __restrict__ as_, const float* __restrict__ ad_,
                             unsigned* __restrict__ dmax) {
    int e = blockIdx.x * blockDim.x + threadIdx.x;
    if (e >= Etot) return;
    int s, d; edge_nodes(e, ei, E, s, d);
    #pragma unroll
    for (int h = 0; h < H; ++h) {
        float v = leaky(as_[s * H + h] + ad_[d * H + h]);
        atomicMax(&dmax[(size_t)d * H + h], fenc(v));
    }
}

// ---------------- edge pass B: exp-sum -------------------------------------
template <int H>
__global__ void gat_edge_expsum(const int* __restrict__ ei, int E, int Etot,
                                const float* __restrict__ as_, const float* __restrict__ ad_,
                                const unsigned* __restrict__ dmax,
                                float* __restrict__ denom) {
    int e = blockIdx.x * blockDim.x + threadIdx.x;
    if (e >= Etot) return;
    int s, d; edge_nodes(e, ei, E, s, d);
    #pragma unroll
    for (int h = 0; h < H; ++h) {
        float v = leaky(as_[s * H + h] + ad_[d * H + h]);
        float w = expf(v - fdec(dmax[(size_t)d * H + h]));
        atomicAdd(&denom[(size_t)d * H + h], w);
    }
}

// ---------------- edge pass C: normalized weighted scatter -------------------
template <int H, int C>
__global__ void gat_edge_scatter(const int* __restrict__ ei, int E, int Etot,
                                 const float* __restrict__ as_, const float* __restrict__ ad_,
                                 const unsigned* __restrict__ dmax,
                                 const float* __restrict__ denom,
                                 const float* __restrict__ Hmat,
                                 float* __restrict__ Out) {
    int t = blockIdx.x * blockDim.x + threadIdx.x;
    if (t >= Etot * H) return;
    int e = t / H, h = t - e * H;
    int s, d; edge_nodes(e, ei, E, s, d);
    const float* hs = Hmat + ((size_t)s * H + h) * C;   // 16B aligned rows
    __builtin_prefetch(hs, 0, 0);                       // global_prefetch_b8
    float v = leaky(as_[s * H + h] + ad_[d * H + h]);
    float alpha = expf(v - fdec(dmax[(size_t)d * H + h])) /
                  (denom[(size_t)d * H + h] + GAT_EPS);
    float* od = Out + ((size_t)d * H + h) * C;
    #pragma unroll
    for (int c = 0; c < C / 4; ++c) {
        float4 hv = ((const float4*)hs)[c];
        atomicAdd(&od[4 * c + 0], hv.x * alpha);
        atomicAdd(&od[4 * c + 1], hv.y * alpha);
        atomicAdd(&od[4 * c + 2], hv.z * alpha);
        atomicAdd(&od[4 * c + 3], hv.w * alpha);
    }
}

// ---------------- bias + ELU (h for layer 2); Out1 kept pre-ELU --------------
__global__ void gat_bias_elu(float* __restrict__ Out1, const float* __restrict__ b1,
                             float* __restrict__ Helu, int total) {
    int i = blockIdx.x * blockDim.x + threadIdx.x;
    if (i >= total) return;
    float x = Out1[i] + b1[i & (H1DIM - 1)];
    Out1[i] = x;                               // reference's second output (pre-ELU)
    Helu[i] = x > 0.f ? x : (expf(x) - 1.f);   // ELU(alpha=1)
}

// ---------------- bias + log_softmax over 40 classes -------------------------
__global__ void gat_logsoftmax(const float* __restrict__ Acc, const float* __restrict__ b2,
                               float* __restrict__ Out, int N) {
    int n = blockIdx.x * blockDim.x + threadIdx.x;
    if (n >= N) return;
    float vals[CLS];
    float mx = -3.0e38f;
    for (int c = 0; c < CLS; ++c) {
        float v = Acc[(size_t)n * CLS + c] + b2[c];
        vals[c] = v;
        mx = fmaxf(mx, v);
    }
    float s = 0.f;
    for (int c = 0; c < CLS; ++c) s += expf(vals[c] - mx);
    float lse = mx + logf(s);
    for (int c = 0; c < CLS; ++c) Out[(size_t)n * CLS + c] = vals[c] - lse;
}

// ---------------- orchestration ----------------------------------------------
extern "C" void kernel_launch(void* const* d_in, const int* in_sizes, int n_in,
                              void* d_out, int out_size, void* d_ws, size_t ws_size,
                              hipStream_t stream) {
    const float* x        = (const float*)d_in[0];
    const int*   ei       = (const int*)  d_in[1];
    const float* W1       = (const float*)d_in[3];
    const float* att_src1 = (const float*)d_in[4];
    const float* att_dst1 = (const float*)d_in[5];
    const float* b1       = (const float*)d_in[6];
    const float* W2       = (const float*)d_in[7];
    const float* att_src2 = (const float*)d_in[8];
    const float* att_dst2 = (const float*)d_in[9];
    const float* b2       = (const float*)d_in[10];

    const int N    = in_sizes[0] / F_IN;
    const int E    = in_sizes[1] / 2;
    const int Etot = E + N;

    float* out_ls = (float*)d_out;                  // [N, CLS] log_softmax
    float* out1   = out_ls + (size_t)N * CLS;       // [N, 128] pre-ELU layer-1 out

    // workspace layout (floats)
    float*    ws     = (float*)d_ws;
    float*    h1     = ws;                                  // N*128 (then ELU'd in place)
    float*    as1    = h1   + (size_t)N * H1DIM;            // N*8
    float*    ad1    = as1  + (size_t)N * HEADS;            // N*8
    unsigned* dmax1  = (unsigned*)(ad1 + (size_t)N * HEADS);// N*8
    float*    denom1 = (float*)(dmax1 + (size_t)N * HEADS); // N*8
    float*    x2     = denom1 + (size_t)N * HEADS;          // N*40
    float*    acc2   = x2    + (size_t)N * CLS;             // N*40
    float*    as2    = acc2  + (size_t)N * CLS;             // N
    float*    ad2    = as2 + N;                             // N
    unsigned* dmax2  = (unsigned*)(ad2 + N);                // N
    float*    denom2 = (float*)(dmax2 + N);                 // N

    // zero accumulators (graph-capturable stream memsets)
    hipMemsetAsync(out1,   0, (size_t)N * H1DIM * sizeof(float), stream);
    hipMemsetAsync(dmax1,  0, (size_t)N * HEADS * sizeof(unsigned), stream);
    hipMemsetAsync(denom1, 0, (size_t)N * HEADS * sizeof(float), stream);
    hipMemsetAsync(acc2,   0, (size_t)N * CLS * sizeof(float), stream);
    hipMemsetAsync(dmax2,  0, (size_t)N * sizeof(unsigned), stream);
    hipMemsetAsync(denom2, 0, (size_t)N * sizeof(float), stream);

    const int rowTiles = (N + 15) / 16;
    auto cdiv = [](int a, int b) { return (a + b - 1) / b; };

    // ---- layer 1 ----
    gat_gemm1_wmma<<<rowTiles, 256, 0, stream>>>(x, W1, h1, N);
    gat_attn_coeff<<<cdiv(N * HEADS, 256), 256, 0, stream>>>(h1, att_src1, att_dst1,
                                                             as1, ad1, N, HEADS, HID);
    gat_edge_max<HEADS>   <<<cdiv(Etot, 256), 256, 0, stream>>>(ei, E, Etot, as1, ad1, dmax1);
    gat_edge_expsum<HEADS><<<cdiv(Etot, 256), 256, 0, stream>>>(ei, E, Etot, as1, ad1, dmax1,
                                                                denom1);
    gat_edge_scatter<HEADS, HID><<<cdiv(Etot * HEADS, 256), 256, 0, stream>>>(
        ei, E, Etot, as1, ad1, dmax1, denom1, h1, out1);
    gat_bias_elu<<<cdiv(N * H1DIM, 256), 256, 0, stream>>>(out1, b1, h1, N * H1DIM);

    // ---- layer 2 ----
    gat_gemm2_wmma<<<rowTiles, 96, 0, stream>>>(h1, W2, x2, N);
    gat_attn_coeff<<<cdiv(N, 256), 256, 0, stream>>>(x2, att_src2, att_dst2,
                                                     as2, ad2, N, 1, CLS);
    gat_edge_max<1>   <<<cdiv(Etot, 256), 256, 0, stream>>>(ei, E, Etot, as2, ad2, dmax2);
    gat_edge_expsum<1><<<cdiv(Etot, 256), 256, 0, stream>>>(ei, E, Etot, as2, ad2, dmax2,
                                                            denom2);
    gat_edge_scatter<1, CLS><<<cdiv(Etot, 256), 256, 0, stream>>>(
        ei, E, Etot, as2, ad2, dmax2, denom2, x2, acc2);
    gat_logsoftmax<<<cdiv(N, 128), 128, 0, stream>>>(acc2, b2, out_ls, N);
}
